// MultiHeadAttention_53558242181713
// MI455X (gfx1250) — matrix operand order repaired and checked by reading, not compiled, not run on hardware
//
#include <hip/hip_runtime.h>

// ---------------- problem constants ----------------
#define B_     16
#define M_     512            // == N in reference
#define HID_   1024
#define HEADS_ 16
#define DH_    64
#define MEM_   1
#define MM_    513            // M + MEM (real softmax length)
#define P_     544            // padded rows: 34 tiles of 16, multiple of 32 for K-loops
#define NT_    34             // P_/16
#define DT_    8              // HID_/128 : d-tiles for the wide value GEMM
#define NEG_F  (-1.0e9f)      // reference mask value; expf(NEG_F) == 0.0f exactly
#define PAD_F  (-3.0e9f)      // padding sentinel (distinct from NEG_F)
#define UNI_F  (1.0f / 513.0f) // uniform weight for fully-masked rows/cols

typedef __bf16 bf16_t;
typedef __attribute__((ext_vector_type(16))) __bf16 v16bf;
typedef __attribute__((ext_vector_type(8)))  __bf16 v8bf;
typedef __attribute__((ext_vector_type(8)))  float  v8f;

// ---------------- helpers ----------------
__device__ inline bf16_t f2bf(float f) {
  unsigned u = __float_as_uint(f);
  unsigned r = (u + 0x7FFFu + ((u >> 16) & 1u)) >> 16;   // round-to-nearest-even
  unsigned short s = (unsigned short)r;
  return __builtin_bit_cast(bf16_t, s);
}

// padded-index mask penalty: 0 = valid, NEG_F = masked, PAD_F = padding
__device__ inline float penalty(int r, const int* mrow) {
  if (r == 0) return 0.0f;          // memory row always valid
  if (r > M_) return PAD_F;         // rows 513..543 are padding
  return mrow[r - 1] ? 0.0f : NEG_F;
}

// A fragment (16x32 bf16): lane holds row (lane&15); lanes<16 K[0..15],
// lanes>=16 K[16..31] contiguous (matches B fragment k order)
__device__ inline v16bf load_a_frag(const bf16_t* rowp, int lane) {
  return *(const v16bf*)(rowp + ((lane >> 4) << 4));
}

// B fragment (32x16 bf16): lane holds column (lane&15); lanes<16 K[0..15],
// lanes>=16 K[16..31] contiguous (same k order as A fragment)
__device__ inline v16bf load_b_frag(const bf16_t* rowp, int lane) {
  return *(const v16bf*)(rowp + ((lane >> 4) << 4));
}

// one 16x16 f32 score tile over K = DH_ = 64 (two bf16 WMMAs)
__device__ inline v8f score_tile(const bf16_t* Xrow, const bf16_t* Yrow, int lane) {
  v8f c = {};
#pragma unroll
  for (int kk = 0; kk < DH_; kk += 32) {
    v16bf a = load_a_frag(Xrow + kk, lane);
    v16bf b = load_b_frag(Yrow + kk, lane);
    c = __builtin_amdgcn_wmma_f32_16x16x32_bf16(false, a, false, b, (short)0, c, false, false);
  }
  return c;
}

// ---------------- kernel 0: build bf16 operands + init sums ----------------
__global__ void k_prep(const float* __restrict__ x, const float* __restrict__ y,
                       const float* __restrict__ xmem, const float* __restrict__ ymem,
                       bf16_t* __restrict__ Xm, bf16_t* __restrict__ Ym,
                       bf16_t* __restrict__ XmT, bf16_t* __restrict__ YmT,
                       float* __restrict__ rowsum, float* __restrict__ colsum) {
  size_t idx = (size_t)blockIdx.x * blockDim.x + threadIdx.x;
  if (idx < (size_t)B_ * P_ * HID_) {
    int d = (int)(idx % HID_);
    size_t t = idx / HID_;
    int r = (int)(t % P_);
    int b = (int)(t / P_);
    float xv = 0.0f, yv = 0.0f;
    if (r == 0)       { xv = xmem[d]; yv = ymem[d]; }
    else if (r <= M_) { xv = x[((size_t)b * M_ + (r - 1)) * HID_ + d];
                        yv = y[((size_t)b * M_ + (r - 1)) * HID_ + d]; }
    Xm [((size_t)b * P_   + r) * HID_ + d] = f2bf(xv);
    Ym [((size_t)b * P_   + r) * HID_ + d] = f2bf(yv);
    XmT[((size_t)b * HID_ + d) * P_   + r] = f2bf(xv);
    YmT[((size_t)b * HID_ + d) * P_   + r] = f2bf(yv);
  }
  if (idx < (size_t)B_ * HEADS_ * P_) {
    rowsum[idx] = 0.0f;
    colsum[idx] = 0.0f;
  }
}

// ---------------- kernel 1: exp-sums (no max pass needed; |aff| <~ 40) --------
__global__ void k_sums(const bf16_t* __restrict__ Xm, const bf16_t* __restrict__ Ym,
                       const int* __restrict__ mask_x, const int* __restrict__ mask_y,
                       float* __restrict__ rowsum, float* __restrict__ colsum) {
  int blk = blockIdx.x;
  int nt = blk % NT_; blk /= NT_;
  int mt = blk % NT_; blk /= NT_;
  int b  = blk;
  int lane = threadIdx.x;
  int m0 = mt * 16, n0 = nt * 16;
  int hi8 = (lane >> 4) << 3;

  float colPen = penalty(n0 + (lane & 15), mask_y + b * M_);
  float rowPen[8];
#pragma unroll
  for (int i = 0; i < 8; ++i) rowPen[i] = penalty(m0 + i + hi8, mask_x + b * M_);

  const bf16_t* Xrow = Xm + ((size_t)b * P_ + m0 + (lane & 15)) * HID_;
  const bf16_t* Yrow = Ym + ((size_t)b * P_ + n0 + (lane & 15)) * HID_;

  for (int h = 0; h < HEADS_; ++h) {
    v8f c = score_tile(Xrow + h * DH_, Yrow + h * DH_, lane);
    size_t sb = ((size_t)b * HEADS_ + h) * P_;
    float colS = 0.0f;
    float rowE[8];
#pragma unroll
    for (int i = 0; i < 8; ++i) {
      float pen = fminf(rowPen[i], colPen);
      float e = (pen < 0.0f) ? 0.0f : __expf(c[i]);   // masked/pad contribute 0
      colS += e;
      rowE[i] = e;
    }
    colS += __shfl_xor(colS, 16, 32);
    if (lane < 16) atomicAdd(&colsum[sb + n0 + lane], colS);
#pragma unroll
    for (int i = 0; i < 8; ++i)
      for (int d = 1; d < 16; d <<= 1)
        rowE[i] += __shfl_xor(rowE[i], d, 32);
    if ((lane & 15) == 0) {
#pragma unroll
      for (int i = 0; i < 8; ++i)
        atomicAdd(&rowsum[sb + m0 + i + hi8], rowE[i]);
    }
  }
}

// ---------------- kernel 2: head-averaged attention maps (bf16) ----------------
// attn_X = softmax over m (per column n); attn_Y = softmax over n (per row m).
// Fully-masked rows/cols -> uniform 1/513 over the 513 real entries (matches ref).
__global__ void k_attn(const bf16_t* __restrict__ Xm, const bf16_t* __restrict__ Ym,
                       const int* __restrict__ mask_x, const int* __restrict__ mask_y,
                       const float* __restrict__ rowsum, const float* __restrict__ colsum,
                       bf16_t* __restrict__ attnXT /*[B][P][P] = attn_X[m][n] at [n][m]*/,
                       bf16_t* __restrict__ attnY  /*[B][P][P] = attn_Y[m][n]*/) {
  int blk = blockIdx.x;
  int nt = blk % NT_; blk /= NT_;
  int mt = blk % NT_; blk /= NT_;
  int b  = blk;
  int lane = threadIdx.x;
  int m0 = mt * 16, n0 = nt * 16;
  int hi8 = (lane >> 4) << 3;

  float colPen = penalty(n0 + (lane & 15), mask_y + b * M_);
  float rowPen[8];
#pragma unroll
  for (int i = 0; i < 8; ++i) rowPen[i] = penalty(m0 + i + hi8, mask_x + b * M_);

  const bf16_t* Xrow = Xm + ((size_t)b * P_ + m0 + (lane & 15)) * HID_;
  const bf16_t* Yrow = Ym + ((size_t)b * P_ + n0 + (lane & 15)) * HID_;

  float axAcc[8] = {0,0,0,0,0,0,0,0};
  float ayAcc[8] = {0,0,0,0,0,0,0,0};

  for (int h = 0; h < HEADS_; ++h) {
    v8f c = score_tile(Xrow + h * DH_, Yrow + h * DH_, lane);
    size_t sb = ((size_t)b * HEADS_ + h) * P_;
    float cs = colsum[sb + n0 + (lane & 15)];
#pragma unroll
    for (int i = 0; i < 8; ++i) {
      float pen = fminf(rowPen[i], colPen);
      float e = (pen < 0.0f) ? 0.0f : __expf(c[i]);
      float rs = rowsum[sb + m0 + i + hi8];
      // attn_X over column n: valid col -> e/colsum; masked col -> uniform over
      // non-pad rows; pad col -> 0 (never read downstream).
      float ax = (colPen == 0.0f) ? (e / cs)
               : ((colPen == NEG_F && rowPen[i] != PAD_F) ? UNI_F : 0.0f);
      // attn_Y over row m: symmetric.
      float ay = (rowPen[i] == 0.0f) ? (e / rs)
               : ((rowPen[i] == NEG_F && colPen != PAD_F) ? UNI_F : 0.0f);
      axAcc[i] += ax;
      ayAcc[i] += ay;
    }
  }
  const float inv = 1.0f / (float)HEADS_;
  int n = n0 + (lane & 15);
#pragma unroll
  for (int i = 0; i < 8; ++i) {
    int m = m0 + i + hi8;
    attnXT[((size_t)b * P_ + n) * P_ + m] = f2bf(axAcc[i] * inv);
    attnY [((size_t)b * P_ + m) * P_ + n] = f2bf(ayAcc[i] * inv);
  }
}

// ---------------- kernels 3/4: attn x values, 16x128 tile per wave ----------------
// C[r][d] = sum_k A[b][r][k] * VT[b][d][k]; store rows MEM..MEM+511 to Out[b][r-MEM][d].
// One A fragment is reused across 8 B fragments (8 independent WMMAs per K-step).
__global__ void k_av(const bf16_t* __restrict__ A /*[B][P][P]*/,
                     const bf16_t* __restrict__ VT /*[B][HID][P]*/,
                     float* __restrict__ Out /*[B][M][HID]*/) {
  int blk = blockIdx.x;
  int dt = blk % DT_; blk /= DT_;
  int rt = blk % NT_; blk /= NT_;
  int b  = blk;
  int lane = threadIdx.x;
  int hi8 = (lane >> 4) << 3;

  const bf16_t* Arow  = A  + ((size_t)b * P_   + rt * 16  + (lane & 15)) * P_;
  const bf16_t* Bbase = VT + ((size_t)b * HID_ + dt * 128 + (lane & 15)) * P_;

  v8f acc[8];
#pragma unroll
  for (int j = 0; j < 8; ++j) acc[j] = {};

  for (int kk = 0; kk < P_; kk += 32) {
    v16bf a = load_a_frag(Arow + kk, lane);
    if (kk + 32 < P_)
      __builtin_prefetch((const void*)(Arow + kk + 32), 0, 3);  // global_prefetch_b8
#pragma unroll
    for (int j = 0; j < 8; ++j) {
      v16bf bm = load_b_frag(Bbase + (size_t)j * 16 * P_ + kk, lane);
      acc[j] = __builtin_amdgcn_wmma_f32_16x16x32_bf16(false, a, false, bm,
                                                       (short)0, acc[j], false, false);
    }
  }
#pragma unroll
  for (int i = 0; i < 8; ++i) {
    int r = rt * 16 + i + hi8;
    if (r >= MEM_ && r < MEM_ + M_) {
      float* o = &Out[((size_t)b * M_ + (r - MEM_)) * HID_ + dt * 128 + (lane & 15)];
#pragma unroll
      for (int j = 0; j < 8; ++j) o[j * 16] = acc[j][i];
    }
  }
}

// ---------------- launch ----------------
extern "C" void kernel_launch(void* const* d_in, const int* in_sizes, int n_in,
                              void* d_out, int out_size, void* d_ws, size_t ws_size,
                              hipStream_t stream) {
  const float* x    = (const float*)d_in[0];
  const float* y    = (const float*)d_in[1];
  const float* xmem = (const float*)d_in[2];
  const float* ymem = (const float*)d_in[3];
  const int* mask_x = (const int*)d_in[4];
  const int* mask_y = (const int*)d_in[5];
  float* out = (float*)d_out;
  float* outX = out;                                   // X_in_Y [B,512,HID]
  float* outY = out + (size_t)B_ * M_ * HID_;          // Y_in_X [B,512,HID]

  char* ws = (char*)d_ws;
  constexpr size_t SZ_MAT  = (size_t)B_ * P_ * HID_ * sizeof(bf16_t);   // 17.8 MB
  constexpr size_t SZ_STAT = (size_t)B_ * HEADS_ * P_ * sizeof(float);  // 0.56 MB
  constexpr size_t SZ_ATT  = (size_t)B_ * P_ * P_ * sizeof(bf16_t);     // 9.5 MB
  bf16_t* Xm     = (bf16_t*)(ws);
  bf16_t* Ym     = (bf16_t*)(ws + SZ_MAT);
  bf16_t* XmT    = (bf16_t*)(ws + 2 * SZ_MAT);
  bf16_t* YmT    = (bf16_t*)(ws + 3 * SZ_MAT);
  float*  rowsum = (float*)(ws + 4 * SZ_MAT);
  float*  colsum = (float*)(ws + 4 * SZ_MAT + SZ_STAT);
  bf16_t* attnXT = (bf16_t*)(ws + 4 * SZ_MAT + 2 * SZ_STAT);
  bf16_t* attnY  = (bf16_t*)(ws + 4 * SZ_MAT + 2 * SZ_STAT + SZ_ATT);

  // 0) prep (bf16 operands + transposes, zero the softmax sums)
  {
    size_t n = (size_t)B_ * P_ * HID_;
    int grid = (int)((n + 255) / 256);
    k_prep<<<grid, 256, 0, stream>>>(x, y, xmem, ymem, Xm, Ym, XmT, YmT,
                                     rowsum, colsum);
  }
  int tileGrid = B_ * NT_ * NT_;   // 18,496 blocks, one wave32 each
  // 1) exp-sums
  k_sums<<<tileGrid, 32, 0, stream>>>(Xm, Ym, mask_x, mask_y, rowsum, colsum);
  // 2) attention maps (head mean, bf16, attn_X stored transposed)
  k_attn<<<tileGrid, 32, 0, stream>>>(Xm, Ym, mask_x, mask_y,
                                      rowsum, colsum, attnXT, attnY);
  // 3) X_in_Y = attn_X^T * Xm   (A = attnXT rows over n, B cols from XmT)
  int avGrid = B_ * NT_ * DT_;     // 4,352 blocks, 16x128 tile each
  k_av<<<avGrid, 32, 0, stream>>>(attnXT, XmT, outX);
  // 4) Y_in_X = attn_Y * Ym
  k_av<<<avGrid, 32, 0, stream>>>(attnY, YmT, outY);
}
